// SupConLoss_14972255994657
// MI455X (gfx1250) — matrix operand order repaired and checked by reading, not compiled, not run on hardware
//
#include <hip/hip_runtime.h>
#include <hip/hip_bf16.h>

typedef __attribute__((ext_vector_type(16))) _Float16 v16h;
typedef __attribute__((ext_vector_type(8)))  _Float16 v8h;
typedef __attribute__((ext_vector_type(8)))  float    v8f;

#define BSZ     4096
#define NVIEWS  2
#define DIM     128
#define NTOT    (BSZ*NVIEWS)                   /* 8192 */
#define NCHUNK  16                             /* column chunks */
#define CT_PER_CHUNK ((NTOT/16)/NCHUNK)        /* 32 col-tiles per chunk */
#define MARGIN  0.1f
#define EPSN    1e-8f
#define FLT_BIG 3.402823466e38f

// ---------------------------------------------------------------------------
// 1) Row-normalize contrast matrix; emit f16 hi/lo split (zn = hi + lo).
//    contrast row i = features[i % 4096, i / 4096, :]
// ---------------------------------------------------------------------------
__global__ __launch_bounds__(256) void sc_normalize(
    const float* __restrict__ feat,
    _Float16* __restrict__ zhi, _Float16* __restrict__ zlo)
{
  const int wave = (blockIdx.x * blockDim.x + threadIdx.x) >> 5;  // row
  const int lane = threadIdx.x & 31;
  const int b = wave & (BSZ - 1);
  const int v = wave >> 12;                       // 4096 = 2^12
  const float4 x = ((const float4*)(feat + (size_t)(b * NVIEWS + v) * DIM))[lane];

  float ss = x.x*x.x + x.y*x.y + x.z*x.z + x.w*x.w;
  #pragma unroll
  for (int o = 16; o >= 1; o >>= 1) ss += __shfl_xor(ss, o, 32);
  const float nrm = fmaxf(sqrtf(ss), EPSN);

  float z[4] = { x.x / nrm, x.y / nrm, x.z / nrm, x.w / nrm };
  const size_t base = (size_t)wave * DIM + lane * 4;
  #pragma unroll
  for (int i = 0; i < 4; ++i) {
    _Float16 h = (_Float16)z[i];
    _Float16 l = (_Float16)(z[i] - (float)h);
    zhi[base + i] = h;
    zlo[base + i] = l;
  }
}

// ---------------------------------------------------------------------------
// 2) Label histogram (labels in [0,100)) -> positive-pair counts.
// ---------------------------------------------------------------------------
__global__ void sc_hist(const int* __restrict__ labels, int* __restrict__ hist)
{
  __shared__ int h[128];
  if (threadIdx.x < 128) h[threadIdx.x] = 0;
  __syncthreads();
  for (int i = threadIdx.x; i < BSZ; i += blockDim.x)
    atomicAdd(&h[labels[i] & 127], 1);
  __syncthreads();
  if (threadIdx.x < 128) hist[threadIdx.x] = h[threadIdx.x];
}

// ---------------------------------------------------------------------------
// 3) Fused sim-GEMM + row reductions.  One wave = 16-row tile x 512-col chunk.
//    sim tile = hiA*hiB + hiA*loB + loA*hiB  (f16-split ~ f32 accuracy),
//    accumulated over K=128 in 4 WMMA K-chunks of 32.
// ---------------------------------------------------------------------------
__global__ __launch_bounds__(256) void sc_simreduce(
    const _Float16* __restrict__ zhi, const _Float16* __restrict__ zlo,
    const int* __restrict__ labels,
    float* __restrict__ psum, float* __restrict__ pmin, float* __restrict__ ppos)
{
  const int lane = threadIdx.x & 31;
  const int wave = (blockIdx.x * blockDim.x + threadIdx.x) >> 5;
  const int rt   = wave >> 4;             // 0..511 row tile
  const int cc   = wave & (NCHUNK - 1);   // 0..15  column chunk
  const int half = lane >> 4;
  const int l15  = lane & 15;

  // labels of the 8 C-tile rows this lane's accumulators map to
  int labRow[8];
  #pragma unroll
  for (int r = 0; r < 8; ++r)
    labRow[r] = labels[(rt * 16 + r + 8 * half) & (BSZ - 1)];

  // Resident A tiles (16x32 f16 per K-chunk, hi + lo), contiguous K order, same as B:
  //   elements 0..7  <- K = kb + 16*half .. +7
  //   elements 8..15 <- K = kb + 16*half + 8 .. +7
  v16h Ahi[4], Alo[4];
  const int rowA = rt * 16 + l15;
  #pragma unroll
  for (int kc = 0; kc < 4; ++kc) {
    const int kb = kc * 32;
    const _Float16* ph = zhi + (size_t)rowA * DIM + kb;
    const _Float16* pl = zlo + (size_t)rowA * DIM + kb;
    v8h h0 = *(const v8h*)(ph + 16 * half);
    v8h h1 = *(const v8h*)(ph + 16 * half + 8);
    v8h l0 = *(const v8h*)(pl + 16 * half);
    v8h l1 = *(const v8h*)(pl + 16 * half + 8);
    #pragma unroll
    for (int i = 0; i < 8; ++i) {
      Ahi[kc][i] = h0[i]; Ahi[kc][i + 8] = h1[i];
      Alo[kc][i] = l0[i]; Alo[kc][i + 8] = l1[i];
    }
  }

  float s[8], m[8], p[8];
  #pragma unroll
  for (int r = 0; r < 8; ++r) { s[r] = 0.f; m[r] = FLT_BIG; p[r] = 0.f; }

  const int ct0 = cc * CT_PER_CHUNK;
  for (int t = 0; t < CT_PER_CHUNK; ++t) {
    const int colB = (ct0 + t) * 16 + l15;           // N = lane & 15
    const int labj = labels[colB & (BSZ - 1)];
    v8f C = {0.f, 0.f, 0.f, 0.f, 0.f, 0.f, 0.f, 0.f};
    #pragma unroll
    for (int kc = 0; kc < 4; ++kc) {
      const int kb = kc * 32;
      // B layout: contiguous K = kb + 16*half .. +15 for column colB
      const _Float16* pbh = zhi + (size_t)colB * DIM + kb + 16 * half;
      const _Float16* pbl = zlo + (size_t)colB * DIM + kb + 16 * half;
      v16h bh = *(const v16h*)pbh;
      v16h bl = *(const v16h*)pbl;
      C = __builtin_amdgcn_wmma_f32_16x16x32_f16(false, Ahi[kc], false, bh, (short)0, C, false, false);
      C = __builtin_amdgcn_wmma_f32_16x16x32_f16(false, Ahi[kc], false, bl, (short)0, C, false, false);
      C = __builtin_amdgcn_wmma_f32_16x16x32_f16(false, Alo[kc], false, bh, (short)0, C, false, false);
    }
    #pragma unroll
    for (int r = 0; r < 8; ++r) {
      const float c = C[r];
      s[r] += c;
      m[r] = fminf(m[r], c);
      p[r] += (labj == labRow[r]) ? c : 0.f;
    }
  }

  // Reduce across the 16 N-lanes of each half-wave (C layout keeps halves apart)
  #pragma unroll
  for (int r = 0; r < 8; ++r) {
    #pragma unroll
    for (int o = 1; o < 16; o <<= 1) {
      s[r] += __shfl_xor(s[r], o, 32);
      m[r] = fminf(m[r], __shfl_xor(m[r], o, 32));
      p[r] += __shfl_xor(p[r], o, 32);
    }
  }
  if (l15 == 0) {
    #pragma unroll
    for (int r = 0; r < 8; ++r) {
      const int row = rt * 16 + r + 8 * half;
      const size_t idx = (size_t)cc * NTOT + row;
      psum[idx] = s[r]; pmin[idx] = m[r]; ppos[idx] = p[r];
    }
  }
}

// ---------------------------------------------------------------------------
// 4) Combine column-chunk partials, compute per-row log term, block-reduce.
// ---------------------------------------------------------------------------
__global__ __launch_bounds__(256) void sc_rowfinal(
    const float* __restrict__ psum, const float* __restrict__ pmin,
    const float* __restrict__ ppos, const int* __restrict__ labels,
    const int* __restrict__ hist, float* __restrict__ blocksum)
{
  __shared__ float sh[256];
  const int row = blockIdx.x * 256 + threadIdx.x;
  float rs = 0.f, rm = FLT_BIG, rp = 0.f;
  #pragma unroll
  for (int cc = 0; cc < NCHUNK; ++cc) {
    const size_t idx = (size_t)cc * NTOT + row;
    rs += psum[idx];
    rm = fminf(rm, pmin[idx]);
    rp += ppos[idx];
  }
  const float pc    = (float)(2 * hist[labels[row & (BSZ - 1)] & 127]);
  const float denom = rs - (float)NTOT * rm + (float)NTOT * MARGIN;
  const float numer = rp - pc * rm + pc * MARGIN;
  sh[threadIdx.x] = __logf(numer / denom) * 0.0f + logf(numer / denom);
  __syncthreads();
  for (int st = 128; st > 0; st >>= 1) {
    if (threadIdx.x < st) sh[threadIdx.x] += sh[threadIdx.x + st];
    __syncthreads();
  }
  if (threadIdx.x == 0) blocksum[blockIdx.x] = sh[0];
}

__global__ void sc_final(const float* __restrict__ blocksum, float* __restrict__ out)
{
  float v = blocksum[threadIdx.x];   // 32 threads, 32 block partials
  #pragma unroll
  for (int o = 16; o >= 1; o >>= 1) v += __shfl_xor(v, o, 32);
  if (threadIdx.x == 0) out[0] = -(v / (float)NTOT);
}

// ---------------------------------------------------------------------------
extern "C" void kernel_launch(void* const* d_in, const int* in_sizes, int n_in,
                              void* d_out, int out_size, void* d_ws, size_t ws_size,
                              hipStream_t stream)
{
  (void)in_sizes; (void)n_in; (void)out_size; (void)ws_size;
  const float* feat   = (const float*)d_in[0];
  const int*   labels = (const int*)d_in[1];
  float*       out    = (float*)d_out;

  char* ws = (char*)d_ws;
  _Float16* zhi  = (_Float16*)(ws);                               // 2 MiB
  _Float16* zlo  = (_Float16*)(ws + (size_t)2097152);             // 2 MiB
  float*    psum = (float*)(ws + (size_t)4194304);                // 512 KiB
  float*    pmin = (float*)(ws + (size_t)4718592);                // 512 KiB
  float*    ppos = (float*)(ws + (size_t)5242880);                // 512 KiB
  int*      hist = (int*)  (ws + (size_t)5767168);                // 512 B
  float*    bsum = (float*)(ws + (size_t)5767680);                // 128 B

  // 8192 rows, 1 wave each -> 1024 blocks of 256
  sc_normalize<<<NTOT / 8, 256, 0, stream>>>(feat, zhi, zlo);
  sc_hist<<<1, 256, 0, stream>>>(labels, hist);
  // 512 row-tiles x 16 col-chunks = 8192 waves -> 1024 blocks of 256
  sc_simreduce<<<(512 * NCHUNK) / 8, 256, 0, stream>>>(zhi, zlo, labels,
                                                       psum, pmin, ppos);
  sc_rowfinal<<<NTOT / 256, 256, 0, stream>>>(psum, pmin, ppos, labels, hist, bsum);
  sc_final<<<1, 32, 0, stream>>>(bsum, out);
}
